// NeuralCDE_19834158972960
// MI455X (gfx1250) — matrix operand order repaired and checked by reading, not compiled, not run on hardware
//
#include <hip/hip_runtime.h>
#include <hip/hip_bf16.h>
#include <math.h>

// ---------- CDNA5 WMMA vector types ----------
typedef __attribute__((ext_vector_type(16))) __bf16 v16bf;
typedef __attribute__((ext_vector_type(8)))  __bf16 v8bf;
typedef __attribute__((ext_vector_type(8)))  float  v8f;

#define B_      256
#define T_      1024
#define TM1_    1023
#define D_      4
#define H_      64
#define W_      128
#define HD_     256
#define ROWS_   16      // batch rows per workgroup (one WMMA M tile)
#define NWG_    16      // B_/ROWS_
#define THREADS_ 256    // 8 wave32 waves

// ---------- dynamic LDS layout (bytes) ----------
#define OFF_WF0   0          // 128*64  bf16 = 16384
#define OFF_WF1   16384      // 128*128 bf16 = 32768
#define OFF_WF2   49152      // 256*128 bf16 = 65536
#define OFF_G0    114688     // 128 f32
#define OFF_G1    115200     // 128 f32
#define OFF_G2    115712     // 256 f32
#define OFF_ABUF  116736     // 16*64  bf16 (A for gemm1)
#define OFF_H1    118784     // 16*128 bf16
#define OFF_H2    122880     // 16*128 bf16
#define OFF_KBUF  126976     // 6*16*64 f32 = 24576 (also init-MLP scratch)
#define OFF_YBUF  151552     // 16*64 f32
#define OFF_DXB   155648     // 64 f32 (also x0 scratch)
#define OFF_CB    155904     // 64 f32
#define OFF_CC    156160     // 64 f32
#define OFF_CD    156416     // 64 f32
#define SMEM_BYTES 156672

// ---------- fast activations: raw v_exp_f32 / v_log_f32 / v_rcp / v_tanh ----
#define LOG2E_ 1.4426950408889634f
#define LN2_   0.6931471805599453f

static __device__ __forceinline__ float fast_exp(float x) {
    return __builtin_amdgcn_exp2f(x * LOG2E_);
}
static __device__ __forceinline__ float fast_sp(float x) {
    // softplus = ln(1 + e^x), branchless select for overflow
    float r = LN2_ * __builtin_amdgcn_logf(1.f + fast_exp(x));
    return (x > 88.f) ? x : r;
}
static __device__ __forceinline__ float fast_tanh(float x) {
#if __has_builtin(__builtin_amdgcn_tanhf)
    return __builtin_amdgcn_tanhf(x);
#else
    // tanh(x) = 1 - 2/(e^{2x}+1); saturates correctly at +-inf of exp2
    float e = __builtin_amdgcn_exp2f(x * (2.f * LOG2E_));
    return 1.f - 2.f * __builtin_amdgcn_rcpf(e + 1.f);
#endif
}

// ---------- fragment loaders (contiguous k order k = half*16 + s, same for A and B) ----------
// 16 consecutive bf16 at p -> one v16bf (VGPR pairs in order)
static __device__ __forceinline__ v16bf ld_frag16(const __bf16* p) {
    v8bf lo = *(const v8bf*)p;
    v8bf hi = *(const v8bf*)(p + 8);
    return __builtin_shufflevector(lo, hi,
        0,1,2,3,4,5,6,7,8,9,10,11,12,13,14,15);
}
// A fragment: row-major [16][K] bf16; lane holds row M=lane&15, 16
// consecutive K values starting at k0 + half*16.
template<int K>
static __device__ __forceinline__ v16bf ld_afrag(const __bf16* Ab, int lane, int k0) {
    const int half = (lane >> 4) & 1;
    const int m    = lane & 15;
    v8bf lo = *(const v8bf*)(Ab + m * K + k0 + half * 16);
    v8bf hi = *(const v8bf*)(Ab + m * K + k0 + half * 16 + 8);
    return __builtin_shufflevector(lo, hi,
        0,1,2,3,4,5,6,7,8,9,10,11,12,13,14,15);
}

static __device__ __forceinline__ v8f wmma_bf16(v16bf a, v16bf b, v8f c) {
    return __builtin_amdgcn_wmma_f32_16x16x32_bf16(
        false, a, false, b, (short)0, c, false, false);
}

__global__ void __launch_bounds__(THREADS_)
neural_cde_persistent(const float* __restrict__ ts,
                      const float* __restrict__ coef_d,
                      const float* __restrict__ coef_c,
                      const float* __restrict__ coef_b,
                      const float* __restrict__ coef_a,
                      const float* __restrict__ W0, const float* __restrict__ b0,
                      const float* __restrict__ W1, const float* __restrict__ b1,
                      const float* __restrict__ W2, const float* __restrict__ b2,
                      const float* __restrict__ F0, const float* __restrict__ g0,
                      const float* __restrict__ F1, const float* __restrict__ g1,
                      const float* __restrict__ F2, const float* __restrict__ g2,
                      const float* __restrict__ L,  const float* __restrict__ lb,
                      float* __restrict__ out) {
    extern __shared__ char smem[];
    __bf16* wF0  = (__bf16*)(smem + OFF_WF0);
    __bf16* wF1  = (__bf16*)(smem + OFF_WF1);
    __bf16* wF2  = (__bf16*)(smem + OFF_WF2);
    float*  g0s  = (float*)(smem + OFF_G0);
    float*  g1s  = (float*)(smem + OFF_G1);
    float*  g2s  = (float*)(smem + OFF_G2);
    __bf16* Abuf = (__bf16*)(smem + OFF_ABUF);
    __bf16* h1b  = (__bf16*)(smem + OFF_H1);
    __bf16* h2b  = (__bf16*)(smem + OFF_H2);
    float*  kb   = (float*)(smem + OFF_KBUF);   // 6144 f32
    float*  yb   = (float*)(smem + OFF_YBUF);   // 1024 f32
    float*  dxb  = (float*)(smem + OFF_DXB);    // 64 f32
    float*  cBv  = (float*)(smem + OFF_CB);
    float*  cCv  = (float*)(smem + OFF_CC);
    float*  cDv  = (float*)(smem + OFF_CD);

    const int tid  = threadIdx.x;
    const int lane = tid & 31;
    const int wave = tid >> 5;
    const int half = (lane >> 4) & 1;
    const int l15  = lane & 15;
    const int r0   = blockIdx.x * ROWS_;   // first batch row of this WG

    // ---- one-time: weights -> bf16 LDS, biases -> f32 LDS ----
    for (int i = tid; i < W_ * H_;   i += THREADS_) wF0[i] = (__bf16)F0[i];
    for (int i = tid; i < W_ * W_;   i += THREADS_) wF1[i] = (__bf16)F1[i];
    for (int i = tid; i < HD_ * W_;  i += THREADS_) wF2[i] = (__bf16)F2[i];
    for (int i = tid; i < W_;  i += THREADS_) { g0s[i] = g0[i]; g1s[i] = g1[i]; }
    for (int i = tid; i < HD_; i += THREADS_) g2s[i] = g2[i];

    // ---- one-time init MLP: y0 = W2 sp(W1 sp(W0 x0 + b0) + b1) + b2 ----
    if (tid < ROWS_ * D_) {     // x0 = coef_a[:, 0, :]
        int row = tid >> 2, d = tid & 3;
        dxb[tid] = coef_a[((size_t)(r0 + row) * TM1_) * D_ + d];
    }
    __syncthreads();
    float* h1i = kb;            // 16*128 f32 scratch
    float* h2i = kb + 2048;     // 16*128 f32 scratch
    for (int idx = tid; idx < ROWS_ * W_; idx += THREADS_) {
        int m = idx >> 7, n = idx & 127;
        float s = b0[n];
#pragma unroll
        for (int d = 0; d < D_; ++d) s += dxb[m * D_ + d] * W0[n * D_ + d];
        h1i[idx] = fast_sp(s);
    }
    __syncthreads();
    for (int idx = tid; idx < ROWS_ * W_; idx += THREADS_) {
        int m = idx >> 7, n = idx & 127;
        float s = b1[n];
        for (int k = 0; k < W_; ++k) s += h1i[m * W_ + k] * W1[n * W_ + k];
        h2i[idx] = fast_sp(s);
    }
    __syncthreads();
    for (int idx = tid; idx < ROWS_ * H_; idx += THREADS_) {
        int m = idx >> 6, n = idx & 63;
        float s = b2[n];
        for (int k = 0; k < W_; ++k) s += h2i[m * W_ + k] * W2[n * W_ + k];
        yb[idx] = s;
    }
    __syncthreads();
    if (tid < ROWS_ * 3) {      // out[:, 0, :] = y0 @ L.T + lb
        int b = tid / 3, o = tid % 3;
        float a = lb[o];
        for (int h = 0; h < H_; ++h) a += yb[b * H_ + h] * L[o * H_ + h];
        out[((size_t)(r0 + b) * T_) * 3 + o] = a;
    }
    __syncthreads();

    // ---- hoist loop-invariant weight B-fragments into VGPRs ----
    // B fragment: lane holds column n = ntile*16 + (lane&15); 16 consecutive
    // K values starting at k0 + half*16, gathered from row-major [n][K].
    const int n0 = wave * 16 + l15;          // gemm1/2 tile col, gemm3 tile-a
    v16bf bF0r[2], bF1r[4], bF2ar[4], bF2br[4];
#pragma unroll
    for (int kw = 0; kw < 2; ++kw)
        bF0r[kw] = ld_frag16(wF0 + n0 * H_ + kw * 32 + half * 16);
#pragma unroll
    for (int kw = 0; kw < 4; ++kw)
        bF1r[kw] = ld_frag16(wF1 + n0 * W_ + kw * 32 + half * 16);
#pragma unroll
    for (int kw = 0; kw < 4; ++kw) {
        bF2ar[kw] = ld_frag16(wF2 + n0 * W_ + kw * 32 + half * 16);
        bF2br[kw] = ld_frag16(wF2 + (n0 + 128) * W_ + kw * 32 + half * 16);
    }
    // loop-invariant per-lane biases
    const float gb0  = g0s[n0];
    const float gb1  = g1s[n0];
    const float gb2a = g2s[n0];
    const float gb2b = g2s[n0 + 128];
    const int   dloc = l15 & 3;              // D index of this lane's column
    const int   ha   = n0 >> 2;              // H index (tile a); tile b = +32

    // ---- persistent scan over 1023 RK steps ----
    for (int t = 0; t < TM1_; ++t) {
        const float dt = ts[t + 1] - ts[t];
        if (tid < ROWS_ * D_) {
            int row = tid >> 2, d = tid & 3;
            size_t base = ((size_t)(r0 + row) * TM1_ + t) * D_ + d;
            cBv[tid] = coef_b[base];
            cCv[tid] = coef_c[base];
            cDv[tid] = coef_d[base];
            if (t + 1 < TM1_) {                 // global_prefetch_b8 next step
                __builtin_prefetch(coef_b + base + D_, 0, 1);
                __builtin_prefetch(coef_c + base + D_, 0, 1);
                __builtin_prefetch(coef_d + base + D_, 0, 1);
            }
        }

        for (int s = 0; s < 6; ++s) {
            // ---- build stage state ys (f32) -> bf16 A fragment buffer ----
            for (int idx = tid; idx < ROWS_ * H_; idx += THREADS_) {
                float a = 0.f;
                switch (s) {
                case 1: a = 0.161f * kb[idx]; break;
                case 2: a = -0.008480655492356989f * kb[idx]
                          +  0.335480655492357f    * kb[1024 + idx]; break;
                case 3: a =  2.8971530571054935f * kb[idx]
                          -  6.359448489975075f  * kb[1024 + idx]
                          +  4.3622954328695815f * kb[2048 + idx]; break;
                case 4: a =  5.325864828439257f   * kb[idx]
                          - 11.748883564062828f   * kb[1024 + idx]
                          +  7.4955393428898365f  * kb[2048 + idx]
                          -  0.09249506636175525f * kb[3072 + idx]; break;
                case 5: a =  5.86145544294642f     * kb[idx]
                          - 12.92096931784711f     * kb[1024 + idx]
                          +  8.159367898576159f    * kb[2048 + idx]
                          -  0.071584973281401f    * kb[3072 + idx]
                          -  0.028269050394068383f * kb[4096 + idx]; break;
                default: break;
                }
                Abuf[idx] = (__bf16)(yb[idx] + dt * a);
            }
            if (tid < ROWS_ * D_) {             // dx(frac) for this stage
                float fr;
                switch (s) {
                case 0: fr = 0.f; break;
                case 1: fr = 0.161f * dt; break;
                case 2: fr = 0.327f * dt; break;
                case 3: fr = 0.9f * dt; break;
                case 4: fr = 0.9800255409045097f * dt; break;
                default: fr = dt; break;
                }
                dxb[tid] = cBv[tid] + fr * (2.f * cCv[tid] + 3.f * fr * cDv[tid]);
            }
            __syncthreads();

            // ---- GEMM1: (16x64)x(64x128) + softplus -> h1b ----
            {
                v8f c = {0.f,0.f,0.f,0.f,0.f,0.f,0.f,0.f};
#pragma unroll
                for (int kw = 0; kw < 2; ++kw)
                    c = wmma_bf16(ld_afrag<H_>(Abuf, lane, kw * 32), bF0r[kw], c);
#pragma unroll
                for (int j = 0; j < 8; ++j) {
                    int m = j + 8 * half;
                    h1b[m * W_ + n0] = (__bf16)fast_sp(c[j] + gb0);
                }
            }
            __syncthreads();

            // ---- GEMM2: (16x128)x(128x128) + softplus -> h2b ----
            {
                v8f c = {0.f,0.f,0.f,0.f,0.f,0.f,0.f,0.f};
#pragma unroll
                for (int kw = 0; kw < 4; ++kw)
                    c = wmma_bf16(ld_afrag<W_>(h1b, lane, kw * 32), bF1r[kw], c);
#pragma unroll
                for (int j = 0; j < 8; ++j) {
                    int m = j + 8 * half;
                    h2b[m * W_ + n0] = (__bf16)fast_sp(c[j] + gb1);
                }
            }
            __syncthreads();

            // ---- GEMM3: (16x128)x(128x256); A frags shared by both N tiles;
            //      tanh, fold dx, shuffle-reduce over D=4 lanes -> kb[s] ----
            {
                v8f ca = {0.f,0.f,0.f,0.f,0.f,0.f,0.f,0.f};
                v8f cb = {0.f,0.f,0.f,0.f,0.f,0.f,0.f,0.f};
#pragma unroll
                for (int kw = 0; kw < 4; ++kw) {
                    v16bf a = ld_afrag<W_>(h2b, lane, kw * 32);
                    ca = wmma_bf16(a, bF2ar[kw], ca);
                    cb = wmma_bf16(a, bF2br[kw], cb);
                }
#pragma unroll
                for (int j = 0; j < 8; ++j) {
                    int m = j + 8 * half;
                    float mul = dxb[m * D_ + dloc];
                    float va = fast_tanh(ca[j] + gb2a) * mul;
                    float vb = fast_tanh(cb[j] + gb2b) * mul;
                    // sum over the 4 lanes holding d = 0..3 of the same h
                    va += __shfl_xor(va, 1, 32);
                    va += __shfl_xor(va, 2, 32);
                    vb += __shfl_xor(vb, 1, 32);
                    vb += __shfl_xor(vb, 2, 32);
                    if (dloc == 0) {
                        kb[s * 1024 + m * H_ + ha]      = va;
                        kb[s * 1024 + m * H_ + ha + 32] = vb;
                    }
                }
            }
            __syncthreads();
        }

        // ---- y update with Butcher weights ----
        for (int idx = tid; idx < ROWS_ * H_; idx += THREADS_) {
            yb[idx] += dt * (0.09646076681806523f * kb[idx]
                           + 0.01f                * kb[1024 + idx]
                           + 0.4798896504144996f  * kb[2048 + idx]
                           + 1.379008574103742f   * kb[3072 + idx]
                           - 3.290069515436081f   * kb[4096 + idx]
                           + 2.324710524099774f   * kb[5120 + idx]);
        }
        __syncthreads();

        // ---- out[:, t+1, :] = y @ L.T + lb ----
        if (tid < ROWS_ * 3) {
            int b = tid / 3, o = tid % 3;
            float a = lb[o];
            for (int h = 0; h < H_; ++h) a += yb[b * H_ + h] * L[o * H_ + h];
            out[((size_t)(r0 + b) * T_ + (t + 1)) * 3 + o] = a;
        }
        // no trailing barrier needed: next phases touch disjoint LDS regions,
        // and the post-build __syncthreads orders everything before GEMM1.
    }
}

extern "C" void kernel_launch(void* const* d_in, const int* in_sizes, int n_in,
                              void* d_out, int out_size, void* d_ws, size_t ws_size,
                              hipStream_t stream) {
    (void)in_sizes; (void)n_in; (void)d_ws; (void)ws_size; (void)out_size;
    const float* ts     = (const float*)d_in[0];
    const float* coef_d = (const float*)d_in[1];
    const float* coef_c = (const float*)d_in[2];
    const float* coef_b = (const float*)d_in[3];
    const float* coef_a = (const float*)d_in[4];
    const float* W0 = (const float*)d_in[5];
    const float* b0 = (const float*)d_in[6];
    const float* W1 = (const float*)d_in[7];
    const float* b1 = (const float*)d_in[8];
    const float* W2 = (const float*)d_in[9];
    const float* b2 = (const float*)d_in[10];
    const float* F0 = (const float*)d_in[11];
    const float* g0 = (const float*)d_in[12];
    const float* F1 = (const float*)d_in[13];
    const float* g1 = (const float*)d_in[14];
    const float* F2 = (const float*)d_in[15];
    const float* g2 = (const float*)d_in[16];
    const float* L  = (const float*)d_in[17];
    const float* lb = (const float*)d_in[18];
    float* out = (float*)d_out;

    hipFuncSetAttribute((const void*)neural_cde_persistent,
                        hipFuncAttributeMaxDynamicSharedMemorySize, SMEM_BYTES);
    neural_cde_persistent<<<dim3(NWG_), dim3(THREADS_), SMEM_BYTES, stream>>>(
        ts, coef_d, coef_c, coef_b, coef_a,
        W0, b0, W1, b1, W2, b2,
        F0, g0, F1, g1, F2, g2, L, lb, out);
}
